// CDSQN_36799279792537
// MI455X (gfx1250) — matrix operand order repaired and checked by reading, not compiled, not run on hardware
//
#include <hip/hip_runtime.h>
#include <hip/hip_bf16.h>

// ---------------- constants (match reference) ----------------
#define BGR    32          // graphs
#define N_PER  1000
#define TOTAL  (BGR * N_PER)   // 32000
#define NEDGE  512000
#define F_IN   128
#define HID    128
#define DD     64
#define NH     3
#define NCOL1  (NH * N_PER * DD)   // 192000  columns of Wg1
#define NCOL2  (NH * DD * DD)      // 12288
#define NCOL3  (NH * DD)           // 192

typedef __attribute__((ext_vector_type(16))) _Float16 v16h;
typedef __attribute__((ext_vector_type(8)))  float    v8f;

__device__ __forceinline__ v8f wmma_f16(v16h a, v16h b, v8f c) {
    // D = A(16x32 f16) * B(32x16 f16) + C(16x16 f32)
    return __builtin_amdgcn_wmma_f32_16x16x32_f16(
        /*neg_a=*/false, a, /*neg_b=*/false, b,
        /*c_mod=*/(short)0, c, /*reuse_a=*/false, /*reuse_b=*/false);
}

// K index inside a 16x32 f16 A operand for half-slot j (0..15), lane group g (0/1)
// per CDNA5 ISA 7.12.2: VGPR v in 0..3 -> K = 2v + 8g (+lo/hi); v in 4..7 -> K = 16 + 2(v-4) + 8g
// For fixed g this is two contiguous 8-half (16B) runs -> two global_load_b128 per operand.
__device__ __forceinline__ int a_kof(int j, int g) {
    int v = j >> 1;
    int base = (v < 4) ? (2 * v) : (16 + 2 * (v - 4));
    return base + 8 * g + (j & 1);
}

__device__ __forceinline__ float softplusf(float x) {
    return (x > 20.f) ? x : log1pf(__expf(x));
}

// ---------------- edge preprocessing ----------------
__global__ void k_deg_init(float* deg) {
    int i = blockIdx.x * blockDim.x + threadIdx.x;
    if (i < TOTAL) deg[i] = 1.0f;                 // self loop
}

__global__ void k_edge_prep(const long long* __restrict__ eidx,
                            int* __restrict__ src32, int* __restrict__ dst32,
                            float* __restrict__ deg) {
    int e = blockIdx.x * blockDim.x + threadIdx.x;
    if (e >= NEDGE) return;
    int s = (int)eidx[e];
    int d = (int)eidx[NEDGE + e];
    src32[e] = s; dst32[e] = d;
    atomicAdd(&deg[d], 1.0f);
}

__global__ void k_dinv(const float* __restrict__ deg, float* __restrict__ dinv) {
    int i = blockIdx.x * blockDim.x + threadIdx.x;
    if (i < TOTAL) dinv[i] = rsqrtf(deg[i]);      // deg >= 1 always
}

__global__ void k_norm(const int* __restrict__ s, const int* __restrict__ d,
                       const float* __restrict__ dinv, float* __restrict__ nrm) {
    int e = blockIdx.x * blockDim.x + threadIdx.x;
    if (e < NEDGE) nrm[e] = dinv[s[e]] * dinv[d[e]];
}

// transpose+convert a 128x128 fp32 weight (in,out) -> fp16 (out,in)
__global__ void k_wcvt128(const float* __restrict__ W, _Float16* __restrict__ Wt) {
    int i = blockIdx.x * blockDim.x + threadIdx.x;   // 16384
    if (i >= HID * HID) return;
    int o = i >> 7, k = i & 127;
    Wt[o * HID + k] = (_Float16)W[k * HID + o];
}

// fused (optional relu) + fp32 -> fp16 conversion of activations
__global__ void k_cvt16(const float* __restrict__ in, _Float16* __restrict__ out, int relu) {
    int i = blockIdx.x * blockDim.x + threadIdx.x;   // TOTAL*128
    if (i >= TOTAL * HID) return;
    float x = in[i];
    if (relu) x = fmaxf(x, 0.f);
    out[i] = (_Float16)x;
}

// ---------------- dense GEMM: out[32000,128] = Ah[32000,128](f16) @ Wt^T ----------------
// Both operands f16 row-major; inner loop is pure b128 loads + wmma (no VALU convert).
// grid = 2000 M-tiles, block = 256 (8 waves = 8 N-tiles). No divergence -> EXEC all ones.
__global__ __launch_bounds__(256)
void k_gemm(const _Float16* __restrict__ Ah, const _Float16* __restrict__ Wt,
            float* __restrict__ Out) {
    int mt   = blockIdx.x;
    int ot   = threadIdx.x >> 5;          // wave id == N tile
    int lane = threadIdx.x & 31;
    int nl   = lane & 15;
    int g    = lane >> 4;

    const _Float16* arow = Ah + (size_t)(mt * 16 + nl) * HID;
    const _Float16* brow = Wt + (size_t)(ot * 16 + nl) * HID;

    v8f acc = {};
    #pragma unroll
    for (int kk = 0; kk < HID; kk += 32) {
        v16h a, b;
        #pragma unroll
        for (int j = 0; j < 16; ++j)
            a[j] = arow[kk + a_kof(j, g)];  // two contiguous 16B runs -> 2x b128
        #pragma unroll
        for (int j = 0; j < 16; ++j)
            b[j] = brow[kk + a_kof(j, g)];  // B: same K order as A (two contiguous 16B runs)
        acc = wmma_f16(a, b, acc);
    }
    float* o0 = Out + (size_t)(mt * 16) * HID + ot * 16 + nl;
    #pragma unroll
    for (int r = 0; r < 8; ++r)
        o0[(size_t)(r + 8 * g) * HID] = acc[r];
}

// out[i,f] = bias[f] + dinv[i]^2 * hw[i,f]   (self-loop + bias init)
__global__ void k_agg_init(const float* __restrict__ hw, const float* __restrict__ dinv,
                           const float* __restrict__ bias, float* __restrict__ out) {
    int i = blockIdx.x * blockDim.x + threadIdx.x;   // TOTAL*128
    if (i >= TOTAL * HID) return;
    int node = i >> 7, f = i & 127;
    float dv = dinv[node];
    out[i] = bias[f] + dv * dv * hw[i];
}

// per-edge scatter: out[dst,:] += norm * hw[src,:]   (32 threads/edge, 4 floats each)
__global__ void k_agg_edges(const float* __restrict__ hw, const int* __restrict__ src,
                            const int* __restrict__ dst, const float* __restrict__ nrm,
                            float* __restrict__ out) {
    int t = blockIdx.x * blockDim.x + threadIdx.x;
    int e = t >> 5;
    if (e >= NEDGE) return;
    int q = (t & 31) * 4;
    float nm = nrm[e];
    const float* hs = hw + (size_t)src[e] * HID + q;
    float*       od = out + (size_t)dst[e] * HID + q;
    #pragma unroll
    for (int k = 0; k < 4; ++k) atomicAdd(od + k, nm * hs[k]);
}

// mean pool per graph (nodes contiguous) -> h_g fp32 + fp16 copy
__global__ __launch_bounds__(128)
void k_pool(const float* __restrict__ h, float* __restrict__ hg, _Float16* __restrict__ hg16) {
    int b = blockIdx.x, f = threadIdx.x;
    const float* p = h + (size_t)b * N_PER * HID + f;
    float s = 0.f;
    for (int n = 0; n < N_PER; ++n) s += p[(size_t)n * HID];
    s *= (1.0f / (float)N_PER);
    hg[b * HID + f] = s;
    hg16[b * HID + f] = (_Float16)s;
}

__global__ void k_zero(float* p, int n) {
    int i = blockIdx.x * blockDim.x + threadIdx.x;
    if (i < n) p[i] = 0.f;
}

// ---------------- fused hyper GEMM + softplus + action reduce ----------------
// h1raw[b,h,d] = sum_n actions[b,n] * softplus(h_g[b,:] . Wg1[:, (h,n,d)] + bg1)
// One wave per (head, d-block of 16, n-chunk of 40). grid = 3*4*25 = 300.
// HBM-bound (streams 98 MB of fp32 Wg1 once); on-the-fly f16 convert hides under memory.
#define NCHUNK 25
#define CHN    40
__global__ __launch_bounds__(32)
void k_hyper1(const _Float16* __restrict__ hg16, const float* __restrict__ Wg1,
              const float* __restrict__ bg1, const float* __restrict__ actions,
              float* __restrict__ h1raw) {
    __shared__ float sact[BGR * CHN];     // actions tile, 5 KB LDS
    int idx = blockIdx.x;
    int hh  = idx / (4 * NCHUNK);
    int rem = idx % (4 * NCHUNK);
    int db  = rem / NCHUNK;
    int ch  = rem % NCHUNK;
    int n0  = ch * CHN;
    int lane = threadIdx.x;
    int nl = lane & 15, g = lane >> 4;

    for (int t = lane; t < BGR * CHN; t += 32) {   // uniform trip count, no divergence
        int b = t / CHN, nn = t % CHN;
        sact[t] = actions[b * N_PER + n0 + nn];
    }
    __syncthreads();

    // preload A operands: 2 M-tiles (b 0-15 / 16-31) x 4 K-steps
    v16h Aop[2][4];
    #pragma unroll
    for (int mt = 0; mt < 2; ++mt) {
        const _Float16* ar = hg16 + (size_t)(mt * 16 + nl) * HID;
        #pragma unroll
        for (int kk = 0; kk < 4; ++kk)
            #pragma unroll
            for (int j = 0; j < 16; ++j)
                Aop[mt][kk][j] = ar[kk * 32 + a_kof(j, g)];
    }

    int d0 = db * 16;
    v8f acc0 = {}, acc1 = {};
    for (int nn = 0; nn < CHN; ++nn) {
        int n = n0 + nn;
        size_t c0 = ((size_t)(hh * N_PER + n)) * DD + d0;
        const float* wcol = Wg1 + c0 + nl;          // coalesced across 16 lanes
        v16h Bop[4];
        #pragma unroll
        for (int kk = 0; kk < 4; ++kk) {
            if (nn + 1 < CHN)                        // prefetch next column block
                __builtin_prefetch(wcol + (size_t)(kk * 32 + g * 16) * NCOL1 + DD, 0, 0);
            #pragma unroll
            for (int j = 0; j < 16; ++j)
                Bop[kk][j] = (_Float16)wcol[(size_t)(kk * 32 + a_kof(j, g)) * NCOL1];
        }
        v8f s0 = {}, s1 = {};
        #pragma unroll
        for (int kk = 0; kk < 4; ++kk) {
            s0 = wmma_f16(Aop[0][kk], Bop[kk], s0);
            s1 = wmma_f16(Aop[1][kk], Bop[kk], s1);
        }
        float bv = bg1[c0 + nl];
        #pragma unroll
        for (int r = 0; r < 8; ++r) {
            int b0 = r + 8 * g;
            acc0[r] += softplusf(s0[r] + bv) * sact[b0 * CHN + nn];
            int b1 = 16 + r + 8 * g;
            acc1[r] += softplusf(s1[r] + bv) * sact[b1 * CHN + nn];
        }
    }
    #pragma unroll
    for (int r = 0; r < 8; ++r) {
        int b0 = r + 8 * g;
        atomicAdd(&h1raw[(b0 * NH + hh) * DD + d0 + nl], acc0[r]);
        int b1 = 16 + r + 8 * g;
        atomicAdd(&h1raw[(b1 * NH + hh) * DD + d0 + nl], acc1[r]);
    }
}

// small hyper heads: out[b,c] = softplus(h_g[b,:] . Wg[:,c] + bg[c])
__global__ void k_hw_small(const float* __restrict__ hg, const float* __restrict__ Wg,
                           const float* __restrict__ bg, float* __restrict__ out, int ncol) {
    int i = blockIdx.x * blockDim.x + threadIdx.x;
    if (i >= BGR * ncol) return;
    int b = i / ncol, c = i % ncol;
    const float* a = hg + b * HID;
    float s = bg[c];
    for (int k = 0; k < HID; ++k) s += a[k] * Wg[(size_t)k * ncol + c];
    out[i] = softplusf(s);
}

// final: SafeSqrt -> h1@w2 -> SafeLog -> dot w3 -> min over heads
__global__ __launch_bounds__(64)
void k_final(const float* __restrict__ h1raw, const float* __restrict__ w2,
             const float* __restrict__ w3, float* __restrict__ out) {
    int b = blockIdx.x, t = threadIdx.x;          // 64 threads = D
    __shared__ float s1[DD];
    __shared__ float red[DD];
    float best = 3.4e38f;
    for (int hh = 0; hh < NH; ++hh) {
        float v = h1raw[(b * NH + hh) * DD + t];
        s1[t] = sqrtf(fmaxf(v, 0.f) + 1e-6f);
        __syncthreads();
        const float* W = w2 + (size_t)(b * NH + hh) * DD * DD;
        float acc = 0.f;
        for (int d = 0; d < DD; ++d) acc += s1[d] * W[d * DD + t];
        float h2v = log1pf(fmaxf(acc, 0.f));
        red[t] = h2v * w3[(b * NH + hh) * DD + t];
        __syncthreads();
        for (int st = 32; st > 0; st >>= 1) { if (t < st) red[t] += red[t + st]; __syncthreads(); }
        if (t == 0) best = fminf(best, red[0]);
        __syncthreads();
    }
    if (t == 0) out[b] = best;
}

// ---------------- launch ----------------
extern "C" void kernel_launch(void* const* d_in, const int* in_sizes, int n_in,
                              void* d_out, int out_size, void* d_ws, size_t ws_size,
                              hipStream_t stream) {
    (void)in_sizes; (void)n_in; (void)out_size; (void)ws_size;
    const float*     x    = (const float*)d_in[0];
    const long long* eidx = (const long long*)d_in[1];
    // d_in[2] = batch_index (contiguous, implied by layout)
    const float* actions = (const float*)d_in[3];
    const float* Wc1 = (const float*)d_in[4];  const float* bc1 = (const float*)d_in[5];
    const float* Wc2 = (const float*)d_in[6];  const float* bc2 = (const float*)d_in[7];
    const float* Wc3 = (const float*)d_in[8];  const float* bc3 = (const float*)d_in[9];
    const float* Wg1 = (const float*)d_in[10]; const float* bg1 = (const float*)d_in[11];
    const float* Wg2 = (const float*)d_in[12]; const float* bg2 = (const float*)d_in[13];
    const float* Wg3 = (const float*)d_in[14]; const float* bg3 = (const float*)d_in[15];
    float* out = (float*)d_out;

    char* ws = (char*)d_ws;
    size_t off = 0;
    auto alloc = [&](size_t bytes) { void* p = ws + off; off += (bytes + 255) & ~(size_t)255; return p; };
    float*     buf0  = (float*)alloc((size_t)TOTAL * HID * 4);  // 16 MB
    float*     buf1  = (float*)alloc((size_t)TOTAL * HID * 4);  // 16 MB
    _Float16*  bufh  = (_Float16*)alloc((size_t)TOTAL * HID * 2); // 8 MB f16 activations
    int*       src32 = (int*)alloc((size_t)NEDGE * 4);
    int*       dst32 = (int*)alloc((size_t)NEDGE * 4);
    float*     nrm   = (float*)alloc((size_t)NEDGE * 4);
    float*     deg   = (float*)alloc((size_t)TOTAL * 4);
    float*     dinv  = (float*)alloc((size_t)TOTAL * 4);
    _Float16*  WtT1  = (_Float16*)alloc((size_t)HID * HID * 2);
    _Float16*  WtT2  = (_Float16*)alloc((size_t)HID * HID * 2);
    _Float16*  WtT3  = (_Float16*)alloc((size_t)HID * HID * 2);
    float*     hg    = (float*)alloc((size_t)BGR * HID * 4);
    _Float16*  hg16  = (_Float16*)alloc((size_t)BGR * HID * 2);
    float*     h1raw = (float*)alloc((size_t)BGR * NH * DD * 4);
    float*     w2buf = (float*)alloc((size_t)BGR * NCOL2 * 4);
    float*     w3buf = (float*)alloc((size_t)BGR * NCOL3 * 4);

    // edge prep
    k_deg_init<<<(TOTAL + 255) / 256, 256, 0, stream>>>(deg);
    k_edge_prep<<<(NEDGE + 255) / 256, 256, 0, stream>>>(eidx, src32, dst32, deg);
    k_dinv<<<(TOTAL + 255) / 256, 256, 0, stream>>>(deg, dinv);
    k_norm<<<(NEDGE + 255) / 256, 256, 0, stream>>>(src32, dst32, dinv, nrm);

    // weight convert+transpose
    k_wcvt128<<<(HID * HID + 255) / 256, 256, 0, stream>>>(Wc1, WtT1);
    k_wcvt128<<<(HID * HID + 255) / 256, 256, 0, stream>>>(Wc2, WtT2);
    k_wcvt128<<<(HID * HID + 255) / 256, 256, 0, stream>>>(Wc3, WtT3);

    const int gemmGrid = TOTAL / 16;                 // 2000
    const int elemGrid = (TOTAL * HID) / 256;        // 16000
    const int edgGrid  = (NEDGE * 32) / 256;         // 64000

    // layer 1 (input x, no relu)
    k_cvt16<<<elemGrid, 256, 0, stream>>>(x, bufh, 0);
    k_gemm<<<gemmGrid, 256, 0, stream>>>(bufh, WtT1, buf1);
    k_agg_init<<<elemGrid, 256, 0, stream>>>(buf1, dinv, bc1, buf0);
    k_agg_edges<<<edgGrid, 256, 0, stream>>>(buf1, src32, dst32, nrm, buf0);
    // layer 2 (relu fused into f16 convert)
    k_cvt16<<<elemGrid, 256, 0, stream>>>(buf0, bufh, 1);
    k_gemm<<<gemmGrid, 256, 0, stream>>>(bufh, WtT2, buf1);
    k_agg_init<<<elemGrid, 256, 0, stream>>>(buf1, dinv, bc2, buf0);
    k_agg_edges<<<edgGrid, 256, 0, stream>>>(buf1, src32, dst32, nrm, buf0);
    // layer 3 (relu fused into f16 convert; no relu on output)
    k_cvt16<<<elemGrid, 256, 0, stream>>>(buf0, bufh, 1);
    k_gemm<<<gemmGrid, 256, 0, stream>>>(bufh, WtT3, buf1);
    k_agg_init<<<elemGrid, 256, 0, stream>>>(buf1, dinv, bc3, buf0);
    k_agg_edges<<<edgGrid, 256, 0, stream>>>(buf1, src32, dst32, nrm, buf0);

    // pool
    k_pool<<<BGR, 128, 0, stream>>>(buf0, hg, hg16);

    // hyper network
    k_zero<<<(BGR * NH * DD + 255) / 256, 256, 0, stream>>>(h1raw, BGR * NH * DD);
    k_hyper1<<<NH * 4 * NCHUNK, 32, 0, stream>>>(hg16, Wg1, bg1, actions, h1raw);
    k_hw_small<<<(BGR * NCOL2 + 255) / 256, 256, 0, stream>>>(hg, Wg2, bg2, w2buf, NCOL2);
    k_hw_small<<<(BGR * NCOL3 + 255) / 256, 256, 0, stream>>>(hg, Wg3, bg3, w3buf, NCOL3);

    // final q computation
    k_final<<<BGR, 64, 0, stream>>>(h1raw, w2buf, w3buf, out);
}
